// ListOpsCircuit_27144193310729
// MI455X (gfx1250) — matrix operand order repaired and checked by reading, not compiled, not run on hardware
//
#include <hip/hip_runtime.h>
#include <stdint.h>

typedef __attribute__((ext_vector_type(16))) _Float16 v16h;
typedef __attribute__((ext_vector_type(8)))  float    v8f;

#define BATCH 16384
#define NNODE 31
#define NPAD  32
#define NI    10
#define NOPS  4
#define NPASS 4
#define WAVES_PER_BLOCK 4
#define ROWS_PER_WAVE   2
#define ROWS_PER_BLOCK  (WAVES_PER_BLOCK * ROWS_PER_WAVE)   // 8
#define GROUPS_PER_WAVE 4                                   // 2 rows x 2 half-rows of 16 nodes
#define COLS  48                                            // 4 ops x 10 ints, padded to 3 N-tiles

__device__ __forceinline__ int iclamp(int x, int lo, int hi) { return x < lo ? lo : (x > hi ? hi : x); }

__global__ __launch_bounds__(128)
void listops_wmma_kernel(const int* __restrict__ cats,
                         const int* __restrict__ ops,
                         const int* __restrict__ lits,
                         const int* __restrict__ left,
                         const int* __restrict__ right,
                         const uint8_t* __restrict__ mask,
                         const float* __restrict__ op_table,
                         float* __restrict__ out)
{
    // state rows are wave-private -> no barriers anywhere
    __shared__ float state[ROWS_PER_BLOCK][NPAD][NI];                      // 10 KB
    __shared__ float scratch[WAVES_PER_BLOCK][GROUPS_PER_WAVE][16][COLS];  // 48 KB

    const int tid    = threadIdx.x;
    const int wave   = tid >> 5;
    const int lane   = tid & 31;
    const int row0   = blockIdx.x * ROWS_PER_BLOCK;
    const int laneHi = lane >> 4;     // 0 or 1
    const int laneLo = lane & 15;

    // ---- B fragments: T flattened to K=128 x COLS=48 fp16, held in VGPRs ----
    // B k order used here (contiguous): col = lane&15; half h -> K = (lane>>4)*16 + h
    v16h Bf[4][3];
    #pragma unroll
    for (int c = 0; c < 4; ++c) {
        #pragma unroll
        for (int t = 0; t < 3; ++t) {
            v16h bfrag;
            #pragma unroll
            for (int h = 0; h < 16; ++h) {
                int K   = c * 32 + (laneHi << 4) + h;     // flattened (i,j), 0..127
                int col = t * 16 + laneLo;                // op*10 + k, 0..47
                int Kc  = K  < 100 ? K  : 99;             // clamp, select 0 after load
                int cc  = col < 40  ? col : 39;
                int i = Kc / 10, j = Kc - i * 10;
                int o = cc / 10, kk = cc - o * 10;
                float v = op_table[((o * NI + i) * NI + j) * NI + kk];
                v = (K < 100 && col < 40) ? v : 0.0f;
                bfrag[h] = (_Float16)v;
            }
            Bf[c][t] = bfrag;
        }
    }

    // ---- Init state: literal nodes -> one-hot, else zero; node 31 stays all-zero ----
    #pragma unroll
    for (int rr = 0; rr < ROWS_PER_WAVE; ++rr) {
        int lr = wave * ROWS_PER_WAVE + rr;
        int b  = row0 + lr;
        int n  = lane;                                    // one node per lane
        bool in_range = (n < NNODE);
        int gidx = in_range ? (b * NNODE + n) : 0;
        int cat = cats[gidx];
        int msk = mask[gidx];
        int lv  = iclamp(lits[gidx], 0, NI - 1);
        bool lit = in_range && (cat == 0) && (msk != 0);
        #pragma unroll
        for (int k = 0; k < NI; ++k)
            state[lr][n][k] = (lit && lv == k) ? 1.0f : 0.0f;
    }

    for (int p = 0; p < NPASS; ++p) {
        // ===== compute phase: logits for every node against all 4 op tables =====
        #pragma unroll
        for (int g = 0; g < GROUPS_PER_WAVE; ++g) {
            int lr = wave * ROWS_PER_WAVE + (g >> 1);
            int b  = row0 + lr;
            int n  = ((g & 1) << 4) + laneLo;             // node handled by this lane pair
            bool valid = (n < NNODE);
            int gidx = valid ? (b * NNODE + n) : 0;
            // invalid lanes gather node 31 (always all-zero) -> no value selects needed
            int li = valid ? iclamp(left[gidx],  0, NNODE - 1) : (NPAD - 1);
            int ri = valid ? iclamp(right[gidx], 0, NNODE - 1) : (NPAD - 1);
            float l[NI], r[NI];
            #pragma unroll
            for (int k = 0; k < NI; ++k) {
                l[k] = state[lr][li][k];
                r[k] = state[lr][ri][k];
            }
            // A k order used here (contiguous): lanes 0-15 M=0-15 K=0..15; lanes 16-31 K=16..31 (matches B).
            // K0/K1 are compile-time constants -> constant register indexing; one select on laneHi.
            v16h Af[4];
            #pragma unroll
            for (int c = 0; c < 4; ++c) {
                v16h afrag;
                #pragma unroll
                for (int h = 0; h < 16; ++h) {
                    const int K0 = c * 32 + h;                          // laneHi == 0
                    const int K1 = K0 + 16;                             // laneHi == 1
                    float v0 = (K0 < 100) ? l[K0 / 10] * r[K0 % 10] : 0.0f;
                    float v1 = (K1 < 100) ? l[K1 / 10] * r[K1 % 10] : 0.0f;
                    float v  = laneHi ? v1 : v0;
                    afrag[h] = (_Float16)v;
                }
                Af[c] = afrag;
            }
            v8f C0 = {}, C1 = {}, C2 = {};
            #pragma unroll
            for (int c = 0; c < 4; ++c) {
                C0 = __builtin_amdgcn_wmma_f32_16x16x32_f16(false, Af[c], false, Bf[c][0], (short)0, C0, false, false);
                C1 = __builtin_amdgcn_wmma_f32_16x16x32_f16(false, Af[c], false, Bf[c][1], (short)0, C1, false, false);
                C2 = __builtin_amdgcn_wmma_f32_16x16x32_f16(false, Af[c], false, Bf[c][2], (short)0, C2, false, false);
            }
            // C layout: VGPR v, lanes 0-15 -> M=v, lanes 16-31 -> M=v+8; N = lane&15
            #pragma unroll
            for (int v = 0; v < 8; ++v) {
                int M = v + (laneHi << 3);
                scratch[wave][g][M][ 0 + laneLo] = C0[v];
                scratch[wave][g][M][16 + laneLo] = C1[v];
                scratch[wave][g][M][32 + laneLo] = C2[v];
            }
        }
        // ===== update phase: select op's 10 columns, softmax, masked state update =====
        #pragma unroll
        for (int g = 0; g < GROUPS_PER_WAVE; ++g) {
            if (lane < 16) {
                int lr = wave * ROWS_PER_WAVE + (g >> 1);
                int b  = row0 + lr;
                int n  = ((g & 1) << 4) + lane;
                if (n < NNODE) {
                    int gidx = b * NNODE + n;
                    int o = iclamp(ops[gidx], 0, NOPS - 1);
                    float logits[NI];
                    #pragma unroll
                    for (int k = 0; k < NI; ++k)
                        logits[k] = scratch[wave][g][lane][o * NI + k];   // runtime LDS offset: fine
                    int cat = cats[gidx];
                    int msk = mask[gidx];
                    bool is_op = (cat == 1) && (msk != 0);
                    float mx = logits[0];
                    #pragma unroll
                    for (int k = 1; k < NI; ++k) mx = fmaxf(mx, logits[k]);
                    float e[NI], s = 0.0f;
                    #pragma unroll
                    for (int k = 0; k < NI; ++k) { e[k] = __expf(logits[k] - mx); s += e[k]; }
                    float inv = 1.0f / s;
                    if (is_op) {
                        #pragma unroll
                        for (int k = 0; k < NI; ++k) state[lr][n][k] = e[k] * inv;
                    }
                    // final pass: emit root logits for node 0
                    if (p == NPASS - 1 && n == 0) {
                        if (cat == 0) {
                            #pragma unroll
                            for (int k = 0; k < NI; ++k) out[b * NI + k] = state[lr][0][k] * 10.0f;
                        } else {
                            #pragma unroll
                            for (int k = 0; k < NI; ++k) out[b * NI + k] = logits[k];
                        }
                    }
                }
            }
        }
    }
}

extern "C" void kernel_launch(void* const* d_in, const int* in_sizes, int n_in,
                              void* d_out, int out_size, void* d_ws, size_t ws_size,
                              hipStream_t stream)
{
    (void)in_sizes; (void)n_in; (void)out_size; (void)d_ws; (void)ws_size;
    const int*      cats     = (const int*)d_in[0];
    const int*      ops      = (const int*)d_in[1];
    const int*      lits     = (const int*)d_in[2];
    const int*      left     = (const int*)d_in[3];
    const int*      right    = (const int*)d_in[4];
    const uint8_t*  mask     = (const uint8_t*)d_in[5];   // numpy bool = 1 byte
    const float*    op_table = (const float*)d_in[6];
    float*          out      = (float*)d_out;

    dim3 grid(BATCH / ROWS_PER_BLOCK);       // 2048 blocks
    dim3 block(WAVES_PER_BLOCK * 32);        // 128 threads = 4 waves
    listops_wmma_kernel<<<grid, block, 0, stream>>>(cats, ops, lits, left, right, mask, op_table, out);
}
